// GAT_43293270344084
// MI455X (gfx1250) — matrix operand order repaired and checked by reading, not compiled, not run on hardware
//
#include <hip/hip_runtime.h>
#include <hip/hip_bf16.h>
#include <stdint.h>

typedef _Float16 f16;
typedef _Float16 v16h __attribute__((ext_vector_type(16)));
typedef _Float16 v8h  __attribute__((ext_vector_type(8)));
typedef float    v8f  __attribute__((ext_vector_type(8)));
typedef int      v4i  __attribute__((ext_vector_type(4)));

#define GN 4096
#define FIN 512
#define HDIM 64
#define NHEADS 8
#define COUT 16
#define GAT_ALPHA 0.2f

// CDNA5 async global->LDS copy path (guarded: falls back to direct global loads)
#if defined(__has_builtin)
# if __has_builtin(__builtin_amdgcn_global_load_async_to_lds_b128) && \
     __has_builtin(__builtin_amdgcn_s_wait_asynccnt)
#  define HAVE_ASYNC_LDS 1
# endif
#endif
#ifndef HAVE_ASYNC_LDS
# define HAVE_ASYNC_LDS 0
#endif

#if HAVE_ASYNC_LDS
__device__ __forceinline__ void a_cp16(const f16* g, f16* l) {
  __builtin_amdgcn_global_load_async_to_lds_b128(
      (__attribute__((address_space(1))) v4i*)(void*)g,
      (__attribute__((address_space(3))) v4i*)(void*)l, 0, 0);
}
#endif

// padded LDS stride (halves) for staged value rows: 40*2B = 80B -> f-row starts
// cover all 64 LDS banks exactly once across 16 lanes (conflict-free b128 reads)
#define BSTRIDE 40

__device__ __forceinline__ float lrelu(float x) { return x > 0.f ? x : GAT_ALPHA * x; }

// ---------- zero scratch accumulators ----------
__global__ void k_zero(float* __restrict__ p, int n) {
  int i = blockIdx.x * blockDim.x + threadIdx.x;
  if (i < n) p[i] = 0.f;
}

// ---------- pack adjacency int32[N][N] -> bitmask u32[N][N/32] ----------
__global__ void k_pack(const int* __restrict__ adj, uint32_t* __restrict__ m) {
  int w = blockIdx.x * blockDim.x + threadIdx.x;
  if (w >= GN * (GN / 32)) return;
  const int* a = adj + (size_t)w * 32;
  uint32_t bits = 0;
  #pragma unroll
  for (int b = 0; b < 32; ++b) bits |= (a[b] > 0 ? 1u : 0u) << b;
  m[w] = bits;
}

// ---------- f32 -> f16 elementwise ----------
__global__ void k_cvt(const float* __restrict__ in, f16* __restrict__ out, int n) {
  int i = blockIdx.x * blockDim.x + threadIdx.x;
  if (i < n) out[i] = (f16)in[i];
}

// ---------- W_heads [8][512][64] -> Wt [8][64][512] f16 ----------
__global__ void k_wt(const float* __restrict__ W, f16* __restrict__ Wt) {
  int i = blockIdx.x * blockDim.x + threadIdx.x;
  if (i >= NHEADS * FIN * HDIM) return;
  int f = i % HDIM, k = (i / HDIM) % FIN, h = i / (HDIM * FIN);
  Wt[(h * HDIM + f) * FIN + k] = (f16)W[i];
}

// ---------- W_out [512][16] -> Wt2 [16][512] f16 ----------
__global__ void k_wt2(const float* __restrict__ W, f16* __restrict__ Wt) {
  int i = blockIdx.x * blockDim.x + threadIdx.x;
  if (i >= FIN * COUT) return;
  int f = i % COUT, k = i / COUT;
  Wt[f * FIN + k] = (f16)W[i];
}

// ---------- per-row max (one block per row) ----------
__global__ void k_rowmax(const float* __restrict__ in, float* __restrict__ out, int cols) {
  __shared__ float red[256];
  float m = -3.4e38f;
  const float* r = in + (size_t)blockIdx.x * cols;
  for (int i = threadIdx.x; i < cols; i += 256) m = fmaxf(m, r[i]);
  red[threadIdx.x] = m;
  __syncthreads();
  for (int s = 128; s > 0; s >>= 1) {
    if ((int)threadIdx.x < s) red[threadIdx.x] = fmaxf(red[threadIdx.x], red[threadIdx.x + s]);
    __syncthreads();
  }
  if (threadIdx.x == 0) out[blockIdx.x] = red[0];
}

// ---------- layer-1 GEMM: h = x @ W_head, + s1/s2, store h^T f16 ----------
__global__ void __launch_bounds__(128)
k_gemm1(const f16* __restrict__ x16, const f16* __restrict__ Wt,
        const float* __restrict__ a_heads,
        f16* __restrict__ ht, float* __restrict__ s1, float* __restrict__ s2) {
  const int head = blockIdx.y;
  const int wave = threadIdx.x >> 5, lane = threadIdx.x & 31;
  const int rlane = lane & 15, g = lane >> 4;
  const int rowbase = blockIdx.x * 16;
  const int colbase = wave * 16;
  const f16* xa = x16 + (size_t)(rowbase + rlane) * FIN;
  const f16* wb = Wt + (size_t)(head * HDIM + colbase + rlane) * FIN + g * 16;
  v8f acc = {};
  for (int k0 = 0; k0 < FIN; k0 += 32) {
    v8h lo = *(const v8h*)(xa + k0 + g * 16);
    v8h hi = *(const v8h*)(xa + k0 + g * 16 + 8);
    v16h A = __builtin_shufflevector(lo, hi, 0,1,2,3,4,5,6,7,8,9,10,11,12,13,14,15);
    v16h B = *(const v16h*)(wb + k0);
    acc = __builtin_amdgcn_wmma_f32_16x16x32_f16(false, A, false, B, (short)0, acc, false, false);
  }
  const int f = colbase + rlane;
  const float a1v = a_heads[head * 2 * HDIM + f];
  const float a2v = a_heads[head * 2 * HDIM + HDIM + f];
  v8h hv;
  #pragma unroll
  for (int r = 0; r < 8; ++r) {
    float v = acc[r];
    hv[r] = (f16)v;
    int m = rowbase + g * 8 + r;
    atomicAdd(&s1[head * GN + m], v * a1v);
    atomicAdd(&s2[head * GN + m], v * a2v);
  }
  *(v8h*)(ht + (size_t)(head * HDIM + f) * GN + rowbase + g * 8) = hv;
}

// ---------- layer-1 fused masked softmax x V ----------
// grid (GN/64, NHEADS), block 128; each wave: 16 rows x 64 cols.
// Value tiles are identical across the 4 waves -> stage once per block in LDS
// via CDNA5 async global->LDS copies, double-buffered.
__global__ void __launch_bounds__(128)
k_att1(const uint32_t* __restrict__ mask32, const f16* __restrict__ ht,
       const float* __restrict__ s1, const float* __restrict__ s2,
       const float* __restrict__ s2max, f16* __restrict__ hcat) {
  __shared__ float s2s[GN];
#if HAVE_ASYNC_LDS
  __shared__ f16 bstage[2][HDIM * BSTRIDE];
#endif
  const int head = blockIdx.y;
  for (int i = threadIdx.x; i < GN; i += 128) s2s[i] = s2[head * GN + i];
  const int wave = threadIdx.x >> 5, lane = threadIdx.x & 31;
  const int rlane = lane & 15, g = lane >> 4;
  const int rowbase = blockIdx.x * 64 + wave * 16;
  const int row = rowbase + rlane;
  const float s1r = s1[head * GN + row];
  const float mi = lrelu(s1r + s2max[head]);
  const uint32_t* mrow = mask32 + (size_t)row * (GN / 32);
  const f16* hthead = ht + (size_t)head * HDIM * GN;
  v8f acc0 = {}, acc1 = {}, acc2 = {}, acc3 = {};
  float wsum = 0.f;

#if HAVE_ASYNC_LDS
  // each thread copies 32B of one value-row f: thread t -> f = t>>1, half-offset (t&1)*16
  const int cf = threadIdx.x >> 1;
  const int ch = (threadIdx.x & 1) * 16;
  const f16* csrc0 = hthead + (size_t)cf * GN + ch;
  {  // prologue: stage tile 0 into buffer 0
    f16* d = &bstage[0][cf * BSTRIDE + ch];
    a_cp16(csrc0, d);
    a_cp16(csrc0 + 8, d + 8);
  }
#endif
  __syncthreads();

  for (int jt = 0; jt < GN / 32; ++jt) {
    const int j0 = jt * 32;
    const uint32_t mw = mrow[jt];
#if HAVE_ASYNC_LDS
    if (jt + 1 < GN / 32) {  // stage next tile into other buffer, keep it in flight
      f16* d = &bstage[(jt + 1) & 1][cf * BSTRIDE + ch];
      const f16* s = csrc0 + (size_t)(jt + 1) * 32;
      a_cp16(s, d);
      a_cp16(s + 8, d + 8);
      __builtin_amdgcn_s_wait_asynccnt(2);  // tile jt complete; jt+1 outstanding
    } else {
      __builtin_amdgcn_s_wait_asynccnt(0);
    }
    __syncthreads();  // all waves' copies for tile jt visible
#endif
    v16h A;
    #pragma unroll
    for (int h = 0; h < 16; ++h) {
      const int k = h + (g << 4);  // contiguous k = 16g + h (matches B)
      float e = lrelu(s1r + s2s[j0 + k]) - mi;
      float w = ((mw >> k) & 1u) ? __expf(e) : 0.f;
      wsum += w;
      A[h] = (f16)w;
    }
#if HAVE_ASYNC_LDS
    const f16* hb = &bstage[jt & 1][rlane * BSTRIDE + g * 16];
    v8h b0l = *(const v8h*)(hb),                 b0h = *(const v8h*)(hb + 8);
    v8h b1l = *(const v8h*)(hb + 16 * BSTRIDE),  b1h = *(const v8h*)(hb + 16 * BSTRIDE + 8);
    v8h b2l = *(const v8h*)(hb + 32 * BSTRIDE),  b2h = *(const v8h*)(hb + 32 * BSTRIDE + 8);
    v8h b3l = *(const v8h*)(hb + 48 * BSTRIDE),  b3h = *(const v8h*)(hb + 48 * BSTRIDE + 8);
    v16h B0 = __builtin_shufflevector(b0l, b0h, 0,1,2,3,4,5,6,7,8,9,10,11,12,13,14,15);
    v16h B1 = __builtin_shufflevector(b1l, b1h, 0,1,2,3,4,5,6,7,8,9,10,11,12,13,14,15);
    v16h B2 = __builtin_shufflevector(b2l, b2h, 0,1,2,3,4,5,6,7,8,9,10,11,12,13,14,15);
    v16h B3 = __builtin_shufflevector(b3l, b3h, 0,1,2,3,4,5,6,7,8,9,10,11,12,13,14,15);
#else
    const f16* hb = hthead + (size_t)rlane * GN + j0 + g * 16;
    v16h B0 = *(const v16h*)(hb);
    v16h B1 = *(const v16h*)(hb + (size_t)16 * GN);
    v16h B2 = *(const v16h*)(hb + (size_t)32 * GN);
    v16h B3 = *(const v16h*)(hb + (size_t)48 * GN);
#endif
    acc0 = __builtin_amdgcn_wmma_f32_16x16x32_f16(false, A, false, B0, (short)0, acc0, false, false);
    acc1 = __builtin_amdgcn_wmma_f32_16x16x32_f16(false, A, false, B1, (short)0, acc1, false, false);
    acc2 = __builtin_amdgcn_wmma_f32_16x16x32_f16(false, A, false, B2, (short)0, acc2, false, false);
    acc3 = __builtin_amdgcn_wmma_f32_16x16x32_f16(false, A, false, B3, (short)0, acc3, false, false);
#if HAVE_ASYNC_LDS
    __syncthreads();  // done reading this buffer before it is overwritten
#endif
  }
  wsum += __shfl_xor(wsum, 16);  // lane L now holds row (L%16) weight sum
  #pragma unroll
  for (int r = 0; r < 8; ++r) {
    const int m = g * 8 + r;
    float l = __shfl(wsum, m);
    float inv = 1.f / fmaxf(l, 1e-30f);
    f16* drow = hcat + (size_t)(rowbase + m) * FIN + head * HDIM + rlane;
    drow[0]  = (f16)(acc0[r] * inv);
    drow[16] = (f16)(acc1[r] * inv);
    drow[32] = (f16)(acc2[r] * inv);
    drow[48] = (f16)(acc3[r] * inv);
  }
}

// ---------- layer-2 GEMM: h2 = hcat @ W_out, + s1/s2, store h2^T f16 ----------
__global__ void __launch_bounds__(128)
k_gemm2(const f16* __restrict__ hcat, const f16* __restrict__ Wt2,
        const float* __restrict__ a_out,
        f16* __restrict__ ht2, float* __restrict__ s1, float* __restrict__ s2) {
  const int wave = threadIdx.x >> 5, lane = threadIdx.x & 31;
  const int rlane = lane & 15, g = lane >> 4;
  const int rowbase = blockIdx.x * 64 + wave * 16;
  const f16* xa = hcat + (size_t)(rowbase + rlane) * FIN;
  const f16* wb = Wt2 + (size_t)rlane * FIN + g * 16;
  v8f acc = {};
  for (int k0 = 0; k0 < FIN; k0 += 32) {
    v8h lo = *(const v8h*)(xa + k0 + g * 16);
    v8h hi = *(const v8h*)(xa + k0 + g * 16 + 8);
    v16h A = __builtin_shufflevector(lo, hi, 0,1,2,3,4,5,6,7,8,9,10,11,12,13,14,15);
    v16h B = *(const v16h*)(wb + k0);
    acc = __builtin_amdgcn_wmma_f32_16x16x32_f16(false, A, false, B, (short)0, acc, false, false);
  }
  const int f = rlane;
  const float a1v = a_out[f], a2v = a_out[COUT + f];
  v8h hv;
  #pragma unroll
  for (int r = 0; r < 8; ++r) {
    float v = acc[r];
    hv[r] = (f16)v;
    int m = rowbase + g * 8 + r;
    atomicAdd(&s1[m], v * a1v);
    atomicAdd(&s2[m], v * a2v);
  }
  *(v8h*)(ht2 + (size_t)f * GN + rowbase + g * 8) = hv;
}

// ---------- layer-2 fused attention + elu + log_softmax ----------
__global__ void __launch_bounds__(128)
k_att2(const uint32_t* __restrict__ mask32, const f16* __restrict__ ht2,
       const float* __restrict__ s1, const float* __restrict__ s2,
       const float* __restrict__ s2max, float* __restrict__ out) {
  __shared__ float s2s[GN];
  for (int i = threadIdx.x; i < GN; i += 128) s2s[i] = s2[i];
  __syncthreads();
  const int wave = threadIdx.x >> 5, lane = threadIdx.x & 31;
  const int rlane = lane & 15, g = lane >> 4;
  const int rowbase = blockIdx.x * 64 + wave * 16;
  const int row = rowbase + rlane;
  const float s1r = s1[row];
  const float mi = lrelu(s1r + s2max[0]);
  const uint32_t* mrow = mask32 + (size_t)row * (GN / 32);
  v8f acc = {};
  float wsum = 0.f;
  for (int jt = 0; jt < GN / 32; ++jt) {
    const int j0 = jt * 32;
    const uint32_t mw = mrow[jt];
    v16h A;
    #pragma unroll
    for (int h = 0; h < 16; ++h) {
      const int k = h + (g << 4);
      float e = lrelu(s1r + s2s[j0 + k]) - mi;
      float w = ((mw >> k) & 1u) ? __expf(e) : 0.f;
      wsum += w;
      A[h] = (f16)w;
    }
    v16h B = *(const v16h*)(ht2 + (size_t)rlane * GN + j0 + g * 16);
    acc = __builtin_amdgcn_wmma_f32_16x16x32_f16(false, A, false, B, (short)0, acc, false, false);
  }
  wsum += __shfl_xor(wsum, 16);
  #pragma unroll
  for (int r = 0; r < 8; ++r) {
    const int m = g * 8 + r;
    float l = __shfl(wsum, m);
    float o = acc[r] / fmaxf(l, 1e-30f);
    o = o > 0.f ? o : __expf(o) - 1.f;  // elu
    float mx = o;
    #pragma unroll
    for (int d = 1; d < 16; d <<= 1) mx = fmaxf(mx, __shfl_xor(mx, d));
    float ex = __expf(o - mx);
    float sum = ex;
    #pragma unroll
    for (int d = 1; d < 16; d <<= 1) sum += __shfl_xor(sum, d);
    out[(size_t)(rowbase + m) * COUT + rlane] = (o - mx) - __logf(sum);
  }
}

extern "C" void kernel_launch(void* const* d_in, const int* in_sizes, int n_in,
                              void* d_out, int out_size, void* d_ws, size_t ws_size,
                              hipStream_t stream) {
  (void)in_sizes; (void)n_in; (void)out_size; (void)ws_size;
  const float* x       = (const float*)d_in[0];
  const int*   adj     = (const int*)d_in[1];
  const float* W_heads = (const float*)d_in[2];
  const float* a_heads = (const float*)d_in[3];
  const float* W_out   = (const float*)d_in[4];
  const float* a_out   = (const float*)d_in[5];
  float* out = (float*)d_out;

  // ---- carve workspace ----
  char* p = (char*)d_ws;
  size_t off = 0;
  auto carve = [&](size_t bytes) { void* q = p + off; off = (off + bytes + 255) & ~(size_t)255; return q; };
  uint32_t* mask32 = (uint32_t*)carve((size_t)GN * (GN / 32) * 4);          // 2 MB
  f16*      x16    = (f16*)carve((size_t)GN * FIN * 2);                      // 4 MB
  f16*      Wt     = (f16*)carve((size_t)NHEADS * HDIM * FIN * 2);           // 512 KB
  f16*      Wt2    = (f16*)carve((size_t)COUT * FIN * 2);                    // 16 KB
  f16*      ht     = (f16*)carve((size_t)NHEADS * HDIM * GN * 2);            // 4 MB
  f16*      hcat   = (f16*)carve((size_t)GN * FIN * 2);                      // 4 MB
  f16*      ht2    = (f16*)carve((size_t)COUT * GN * 2);                     // 128 KB
  float*    sbuf   = (float*)carve((size_t)(NHEADS * GN * 2 + GN * 2) * 4);  // s1/s2 (L1), s1/s2 (L2)
  float*    s1_all = sbuf;
  float*    s2_all = sbuf + NHEADS * GN;
  float*    s1_2   = sbuf + NHEADS * GN * 2;
  float*    s2_2   = sbuf + NHEADS * GN * 2 + GN;
  float*    S2max  = (float*)carve((NHEADS + 1) * 4);
  float*    S2max2 = S2max + NHEADS;

  const int nS = NHEADS * GN * 2 + GN * 2;
  k_zero<<<(nS + 255) / 256, 256, 0, stream>>>(sbuf, nS);
  k_pack<<<(GN * (GN / 32) + 255) / 256, 256, 0, stream>>>(adj, mask32);
  k_cvt<<<(GN * FIN + 255) / 256, 256, 0, stream>>>(x, x16, GN * FIN);
  k_wt<<<(NHEADS * FIN * HDIM + 255) / 256, 256, 0, stream>>>(W_heads, Wt);
  k_wt2<<<(FIN * COUT + 255) / 256, 256, 0, stream>>>(W_out, Wt2);

  k_gemm1<<<dim3(GN / 16, NHEADS), 128, 0, stream>>>(x16, Wt, a_heads, ht, s1_all, s2_all);
  k_rowmax<<<NHEADS, 256, 0, stream>>>(s2_all, S2max, GN);
  k_att1<<<dim3(GN / 64, NHEADS), 128, 0, stream>>>(mask32, ht, s1_all, s2_all, S2max, hcat);

  k_gemm2<<<GN / 64, 128, 0, stream>>>(hcat, Wt2, a_out, ht2, s1_2, s2_2);
  k_rowmax<<<1, 256, 0, stream>>>(s2_2, S2max2, GN);
  k_att2<<<GN / 64, 128, 0, stream>>>(mask32, ht2, s1_2, s2_2, S2max2, out);
}
